// GRU_1090921693290
// MI455X (gfx1250) — matrix operand order repaired and checked by reading, not compiled, not run on hardware
//
#include <hip/hip_runtime.h>
#include <cstdint>
#include <cstddef>

// ---------------- problem constants ----------------
#define TT    1000
#define BB    16
#define B2    32          // fwd+bwd combined batch
#define DIN   440
#define DINP  448         // 440 padded to multiple of 32
#define HH    512
#define NCLS  2000
#define MROWS (TT*B2)     // 32000 rows for projection GEMMs
#define FCM   (TT*BB)     // 16000 rows for final FC
#define SCAN_WGS 16       // persistent workgroups in the recurrent scan

typedef _Float16 half_t;
typedef __attribute__((ext_vector_type(16))) _Float16     v16h;
typedef __attribute__((ext_vector_type(8)))  float        v8f;
typedef __attribute__((ext_vector_type(4)))  unsigned int u32x4;

union Frag { v16h h; u32x4 q[2]; };

__device__ __forceinline__ v8f wmma_f16(Frag a, Frag b, v8f c) {
  // D(f32 16x16) = A(f16 16x32) * B(f16 32x16) + C
  return __builtin_amdgcn_wmma_f32_16x16x32_f16(false, a.h, false, b.h,
                                                (short)0, c, false, false);
}

__device__ __forceinline__ float sigmoidf_(float x) {
  return 1.0f / (1.0f + __expf(-x));
}

// ---------------- elementwise / staging kernels ----------------

// f32 (rows,K) -> f16 (rows,Kpad) with zero padding of K..Kpad
__global__ void k_cvt_pad(const float* __restrict__ s, half_t* __restrict__ d,
                          int rows, int K, int Kpad) {
  size_t n = (size_t)rows * Kpad;
  for (size_t i = (size_t)blockIdx.x * blockDim.x + threadIdx.x; i < n;
       i += (size_t)gridDim.x * blockDim.x) {
    int k = (int)(i % Kpad);
    size_t r = i / Kpad;
    d[i] = (k < K) ? (half_t)s[r * (size_t)K + k] : (half_t)0.0f;
  }
}

// layer-0 GEMM input: x2[t, b<16] = x[t,b]; x2[t, b>=16] = x[T-1-t, b-16]; K padded
__global__ void k_build_x2_l0(const float* __restrict__ x, half_t* __restrict__ x2) {
  size_t n = (size_t)MROWS * DINP;
  for (size_t i = (size_t)blockIdx.x * blockDim.x + threadIdx.x; i < n;
       i += (size_t)gridDim.x * blockDim.x) {
    int k = (int)(i % DINP);
    size_t r = i / DINP;
    int t = (int)(r / B2), b = (int)(r % B2);
    float v = 0.0f;
    if (k < DIN) {
      if (b < BB) v = x[((size_t)t * BB + b) * DIN + k];
      else        v = x[((size_t)(TT - 1 - t) * BB + (b - BB)) * DIN + k];
    }
    x2[i] = (half_t)v;
  }
}

// layer-1 GEMM input from layer-0 hidden states hs (T, 32, 512) f32.
// x_next[t,b] = concat(hs[t,b], hs[T-1-t, 16+b]); x2 = concat_batch(x_next, flip_t(x_next))
__global__ void k_build_x2_l1(const float* __restrict__ hs, half_t* __restrict__ x2) {
  size_t n = (size_t)MROWS * 1024;
  for (size_t i = (size_t)blockIdx.x * blockDim.x + threadIdx.x; i < n;
       i += (size_t)gridDim.x * blockDim.x) {
    int k = (int)(i % 1024);
    size_t r = i / 1024;
    int t = (int)(r / B2), b = (int)(r % B2);
    int bb = b & 15;
    float v;
    if (b < BB) {
      v = (k < HH) ? hs[((size_t)t * B2 + bb) * HH + k]
                   : hs[((size_t)(TT - 1 - t) * B2 + BB + bb) * HH + (k - HH)];
    } else {
      v = (k < HH) ? hs[((size_t)(TT - 1 - t) * B2 + bb) * HH + k]
                   : hs[((size_t)t * B2 + BB + bb) * HH + (k - HH)];
    }
    x2[i] = (half_t)v;
  }
}

// FC input (T*B, 1024) f16 from layer-1 hidden states
__global__ void k_build_fcin(const float* __restrict__ hs, half_t* __restrict__ fcin) {
  size_t n = (size_t)FCM * 1024;
  for (size_t i = (size_t)blockIdx.x * blockDim.x + threadIdx.x; i < n;
       i += (size_t)gridDim.x * blockDim.x) {
    int k = (int)(i % 1024);
    size_t r = i / 1024;
    int t = (int)(r / BB), b = (int)(r % BB);
    float v = (k < HH) ? hs[((size_t)t * B2 + b) * HH + k]
                       : hs[((size_t)(TT - 1 - t) * B2 + BB + b) * HH + (k - HH)];
    fcin[i] = (half_t)v;
  }
}

__global__ void k_init_bar(unsigned* bar) {
  if (threadIdx.x < 2) bar[threadIdx.x] = 0u;
}

// ---------------- WMMA GEMM: C(M,N) = A(M,K) @ B(N,K)^T + bias ----------------
// A,B f16 row-major (ld = K), C f32 row-major (ld = N).
// Block = 128 threads (4 waves); each wave computes a 32x32 macro-tile.
// M multiple of 32, N multiple of 16, K multiple of 32.
// The (rare) 16-wide tail macro-tile is handled by a hoisted branch so the hot
// k-loop contains no EXEC manipulation around the WMMAs.
__global__ void k_gemm(const half_t* __restrict__ A, const half_t* __restrict__ Bm,
                       const float* __restrict__ bias, float* __restrict__ C,
                       int M, int N, int K) {
  const int wave = threadIdx.x >> 5;
  const int lane = threadIdx.x & 31;
  const int lr = lane & 15;
  const int lh = lane >> 4;
  const int m0 = blockIdx.y * 32;
  const int n0 = (blockIdx.x * 4 + wave) * 32;
  if (n0 >= N) return;
  const bool ok1 = (n0 + 16) < N;
  const int akoff = lh ? 16 : 0;   // A frag: lanes16-31 hold K=16..31
  const int bkoff = lh ? 16 : 0;   // B frag: lanes16-31 hold K=16..31

  const half_t* Ar0 = A + (size_t)(m0 + lr) * K;
  const half_t* Ar1 = A + (size_t)(m0 + 16 + lr) * K;
  const half_t* Br0 = Bm + (size_t)(n0 + lr) * K;

  v8f zro = {0.f, 0.f, 0.f, 0.f, 0.f, 0.f, 0.f, 0.f};
  v8f acc00 = zro, acc10 = zro;
  const float bias0 = bias[n0 + lr];

  if (ok1) {
    // full 32x32 macro-tile: 4 WMMAs per k-step, no branches in loop
    const half_t* Br1 = Bm + (size_t)(n0 + 16 + lr) * K;
    v8f acc01 = zro, acc11 = zro;
    for (int k = 0; k < K; k += 32) {
      Frag a0, a1, b0, b1;
      a0.q[0] = *(const u32x4*)(Ar0 + k + akoff);
      a0.q[1] = *(const u32x4*)(Ar0 + k + akoff + 8);
      a1.q[0] = *(const u32x4*)(Ar1 + k + akoff);
      a1.q[1] = *(const u32x4*)(Ar1 + k + akoff + 8);
      b0.q[0] = *(const u32x4*)(Br0 + k + bkoff);
      b0.q[1] = *(const u32x4*)(Br0 + k + bkoff + 8);
      b1.q[0] = *(const u32x4*)(Br1 + k + bkoff);
      b1.q[1] = *(const u32x4*)(Br1 + k + bkoff + 8);
      acc00 = wmma_f16(a0, b0, acc00);
      acc10 = wmma_f16(a1, b0, acc10);
      acc01 = wmma_f16(a0, b1, acc01);
      acc11 = wmma_f16(a1, b1, acc11);
    }
    const float bias1 = bias[n0 + 16 + lr];
#pragma unroll
    for (int v = 0; v < 8; ++v) {
      const int mrow = m0 + v + 8 * lh;    // C layout: vgpr v -> M = v + 8*(lane>=16)
      C[(size_t)mrow * N + n0 + lr]             = acc00[v] + bias0;
      C[(size_t)(mrow + 16) * N + n0 + lr]      = acc10[v] + bias0;
      C[(size_t)mrow * N + n0 + 16 + lr]        = acc01[v] + bias1;
      C[(size_t)(mrow + 16) * N + n0 + 16 + lr] = acc11[v] + bias1;
    }
  } else {
    // 32x16 tail macro-tile
    for (int k = 0; k < K; k += 32) {
      Frag a0, a1, b0;
      a0.q[0] = *(const u32x4*)(Ar0 + k + akoff);
      a0.q[1] = *(const u32x4*)(Ar0 + k + akoff + 8);
      a1.q[0] = *(const u32x4*)(Ar1 + k + akoff);
      a1.q[1] = *(const u32x4*)(Ar1 + k + akoff + 8);
      b0.q[0] = *(const u32x4*)(Br0 + k + bkoff);
      b0.q[1] = *(const u32x4*)(Br0 + k + bkoff + 8);
      acc00 = wmma_f16(a0, b0, acc00);
      acc10 = wmma_f16(a1, b0, acc10);
    }
#pragma unroll
    for (int v = 0; v < 8; ++v) {
      const int mrow = m0 + v + 8 * lh;
      C[(size_t)mrow * N + n0 + lr]        = acc00[v] + bias0;
      C[(size_t)(mrow + 16) * N + n0 + lr] = acc10[v] + bias0;
    }
  }
}

// ---------------- persistent GRU scan ----------------
// Grid = SCAN_WGS workgroups x 64 threads (2 waves). WG g owns 32 hidden columns.
// Per step: h_slice_new = GRU(h_prev) using 3 WMMA GEMM slices (32x32x512 each),
// then a device-scope sense-reversing barrier. h16 = f16 ping-pong copy of h.
__global__ void k_gru_scan(const float* __restrict__ xz, const float* __restrict__ xh,
                           const float* __restrict__ xr,
                           const half_t* __restrict__ Uz, const half_t* __restrict__ Uh,
                           const half_t* __restrict__ Ur,
                           float* __restrict__ hs, half_t* __restrict__ h16,
                           unsigned* __restrict__ bar) {
  const int g = blockIdx.x;
  const int wave = threadIdx.x >> 5;
  const int lane = threadIdx.x & 31;
  const int lr = lane & 15;
  const int lh = lane >> 4;
  const int ncol0 = g * 32;       // this WG's hidden-column slice
  const int m0 = wave * 16;       // this wave's 16 batch rows
  const int akoff = lh ? 16 : 0;
  const int bkoff = lh ? 16 : 0;

  const half_t* Uzr[2] = { Uz + (size_t)(ncol0 + lr) * HH, Uz + (size_t)(ncol0 + 16 + lr) * HH };
  const half_t* Uhr[2] = { Uh + (size_t)(ncol0 + lr) * HH, Uh + (size_t)(ncol0 + 16 + lr) * HH };
  const half_t* Urr[2] = { Ur + (size_t)(ncol0 + lr) * HH, Ur + (size_t)(ncol0 + 16 + lr) * HH };

  const v8f zro = {0.f, 0.f, 0.f, 0.f, 0.f, 0.f, 0.f, 0.f};

  for (int t = 0; t < TT; ++t) {
    v8f az[2] = {zro, zro}, ah[2] = {zro, zro}, ar[2] = {zro, zro};

    if (t > 0) {
      const half_t* hprev = h16 + ((size_t)((t - 1) & 1)) * (B2 * HH);
      const half_t* Arow = hprev + (size_t)(m0 + lr) * HH;
      for (int k = 0; k < HH; k += 32) {
        Frag a;
        a.q[0] = *(const u32x4*)(Arow + k + akoff);
        a.q[1] = *(const u32x4*)(Arow + k + akoff + 8);
#pragma unroll
        for (int sub = 0; sub < 2; ++sub) {
          Frag b;
          b.q[0] = *(const u32x4*)(Uzr[sub] + k + bkoff);
          b.q[1] = *(const u32x4*)(Uzr[sub] + k + bkoff + 8);
          az[sub] = wmma_f16(a, b, az[sub]);
          b.q[0] = *(const u32x4*)(Uhr[sub] + k + bkoff);
          b.q[1] = *(const u32x4*)(Uhr[sub] + k + bkoff + 8);
          ah[sub] = wmma_f16(a, b, ah[sub]);
          b.q[0] = *(const u32x4*)(Urr[sub] + k + bkoff);
          b.q[1] = *(const u32x4*)(Urr[sub] + k + bkoff + 8);
          ar[sub] = wmma_f16(a, b, ar[sub]);
        }
      }
    }

    half_t* hnext = h16 + ((size_t)(t & 1)) * (B2 * HH);
#pragma unroll
    for (int sub = 0; sub < 2; ++sub) {
      const int n = ncol0 + sub * 16 + lr;
#pragma unroll
      for (int v = 0; v < 8; ++v) {
        const int m = m0 + v + 8 * lh;
        const size_t idx = ((size_t)t * B2 + m) * HH + n;
        const float xzv = xz[idx], xhv = xh[idx], xrv = xr[idx];
        const float hp = (t > 0) ? hs[idx - (size_t)B2 * HH] : 0.0f;
        const float zt = sigmoidf_(xzv + az[sub][v]);
        const float rt = sigmoidf_(xrv + ar[sub][v]);
        const float hc = tanhf(xhv + rt * ah[sub][v]);
        const float hn = zt * hp + (1.0f - zt) * hc;
        hs[idx] = hn;
        hnext[(size_t)m * HH + n] = (half_t)hn;
      }
    }

    // --- device-scope barrier across SCAN_WGS workgroups ---
    __threadfence();
    __syncthreads();
    if (threadIdx.x == 0) {
      unsigned gen = __hip_atomic_load(&bar[1], __ATOMIC_ACQUIRE, __HIP_MEMORY_SCOPE_AGENT);
      unsigned arr = __hip_atomic_fetch_add(&bar[0], 1u, __ATOMIC_ACQ_REL, __HIP_MEMORY_SCOPE_AGENT);
      if (arr == (unsigned)(gridDim.x - 1)) {
        __hip_atomic_store(&bar[0], 0u, __ATOMIC_RELAXED, __HIP_MEMORY_SCOPE_AGENT);
        __hip_atomic_fetch_add(&bar[1], 1u, __ATOMIC_RELEASE, __HIP_MEMORY_SCOPE_AGENT);
      } else {
        while (__hip_atomic_load(&bar[1], __ATOMIC_ACQUIRE, __HIP_MEMORY_SCOPE_AGENT) == gen)
          __builtin_amdgcn_s_sleep(2);
      }
    }
    __syncthreads();
  }
}

// ---------------- log-softmax + per-row NLL / err ----------------
__global__ void k_logsoftmax(float* __restrict__ out, const int* __restrict__ lab,
                             float* __restrict__ nll, float* __restrict__ errf) {
  __shared__ float sv[256];
  __shared__ int si[256];
  const int r = blockIdx.x;
  const int tid = threadIdx.x;
  float* row = out + (size_t)r * NCLS;

  float m = -3.4e38f; int mi = 0;
  for (int i = tid; i < NCLS; i += 256) {
    float v = row[i];
    if (v > m) { m = v; mi = i; }
  }
  sv[tid] = m; si[tid] = mi; __syncthreads();
  for (int s = 128; s > 0; s >>= 1) {
    if (tid < s) {
      float v2 = sv[tid + s]; int i2 = si[tid + s];
      if (v2 > sv[tid] || (v2 == sv[tid] && i2 < si[tid])) { sv[tid] = v2; si[tid] = i2; }
    }
    __syncthreads();
  }
  const float rowmax = sv[0];
  const int pred = si[0];
  __syncthreads();

  float acc = 0.0f;
  for (int i = tid; i < NCLS; i += 256) acc += __expf(row[i] - rowmax);
  sv[tid] = acc; __syncthreads();
  for (int s = 128; s > 0; s >>= 1) {
    if (tid < s) sv[tid] += sv[tid + s];
    __syncthreads();
  }
  const float lse = rowmax + __logf(sv[0]);

  if (tid == 0) {
    const int l = lab[r];
    nll[r] = lse - row[l];                 // read BEFORE in-place update
    errf[r] = (pred != l) ? 1.0f : 0.0f;
  }
  __syncthreads();
  for (int i = tid; i < NCLS; i += 256) row[i] = row[i] - lse;
}

// deterministic final reduction (fixed order, no float atomics)
__global__ void k_reduce(const float* __restrict__ nll, const float* __restrict__ errf,
                         float* __restrict__ dout) {
  __shared__ float s1[256], s2[256];
  float a = 0.0f, b = 0.0f;
  for (int i = threadIdx.x; i < FCM; i += 256) { a += nll[i]; b += errf[i]; }
  s1[threadIdx.x] = a; s2[threadIdx.x] = b; __syncthreads();
  for (int s = 128; s > 0; s >>= 1) {
    if (threadIdx.x < s) { s1[threadIdx.x] += s1[threadIdx.x + s]; s2[threadIdx.x] += s2[threadIdx.x + s]; }
    __syncthreads();
  }
  if (threadIdx.x == 0) {
    dout[0] = s1[0] / (float)FCM;
    dout[1] = s2[0] / (float)FCM;
  }
}

// ---------------- host orchestration ----------------
static inline void launch_gemm(const half_t* A, const half_t* B, const float* bias,
                               float* C, int M, int N, int K, hipStream_t s) {
  int nmac = (N + 31) / 32;
  dim3 grid((nmac + 3) / 4, M / 32);
  k_gemm<<<grid, 128, 0, s>>>(A, B, bias, C, M, N, K);
}

extern "C" void kernel_launch(void* const* d_in, const int* in_sizes, int n_in,
                              void* d_out, int out_size, void* d_ws, size_t ws_size,
                              hipStream_t stream) {
  (void)in_sizes; (void)n_in; (void)out_size; (void)ws_size;
  const float* x = (const float*)d_in[0];
  const int* lab = (const int*)d_in[1];
  float* OUT = (float*)d_out;

  // workspace carve (256B aligned)
  char* p = (char*)d_ws;
  auto take = [&](size_t bytes) -> char* {
    char* r = p;
    p += (bytes + 255) & ~(size_t)255;
    return r;
  };
  half_t*  X2   = (half_t*)take((size_t)MROWS * 1024 * sizeof(half_t));  // layer GEMM input (max K=1024)
  float*   XZ   = (float*)take((size_t)MROWS * HH * sizeof(float));
  float*   XH   = (float*)take((size_t)MROWS * HH * sizeof(float));
  float*   XR   = (float*)take((size_t)MROWS * HH * sizeof(float));
  float*   HS   = (float*)take((size_t)MROWS * HH * sizeof(float));      // hidden states (reused per layer)
  half_t*  W16z = (half_t*)take((size_t)HH * 1024 * sizeof(half_t));
  half_t*  W16h = (half_t*)take((size_t)HH * 1024 * sizeof(half_t));
  half_t*  W16r = (half_t*)take((size_t)HH * 1024 * sizeof(half_t));
  half_t*  U16z = (half_t*)take((size_t)HH * HH * sizeof(half_t));
  half_t*  U16h = (half_t*)take((size_t)HH * HH * sizeof(half_t));
  half_t*  U16r = (half_t*)take((size_t)HH * HH * sizeof(half_t));
  half_t*  FCW  = (half_t*)take((size_t)NCLS * 1024 * sizeof(half_t));
  half_t*  H16  = (half_t*)take((size_t)2 * B2 * HH * sizeof(half_t));
  unsigned* BAR = (unsigned*)take(256);
  float*   NLLb = (float*)take((size_t)FCM * sizeof(float));
  float*   ERRb = (float*)take((size_t)FCM * sizeof(float));
  half_t*  FCIN = X2;   // alias: x2 no longer needed once FC input is built

  for (int l = 0; l < 2; ++l) {
    const int Ksrc = l ? 1024 : DIN;
    const int Kp   = l ? 1024 : DINP;
    if (l == 0) k_build_x2_l0<<<2048, 256, 0, stream>>>(x, X2);
    else        k_build_x2_l1<<<2048, 256, 0, stream>>>(HS, X2);

    const float* Wz = (const float*)d_in[3 + 9 * l];
    const float* bz = (const float*)d_in[4 + 9 * l];
    const float* Wh = (const float*)d_in[5 + 9 * l];
    const float* bh = (const float*)d_in[6 + 9 * l];
    const float* Wr = (const float*)d_in[7 + 9 * l];
    const float* br = (const float*)d_in[8 + 9 * l];
    const float* Uz = (const float*)d_in[9 + 9 * l];
    const float* Uh = (const float*)d_in[10 + 9 * l];
    const float* Ur = (const float*)d_in[11 + 9 * l];

    k_cvt_pad<<<1024, 256, 0, stream>>>(Wz, W16z, HH, Ksrc, Kp);
    k_cvt_pad<<<1024, 256, 0, stream>>>(Wh, W16h, HH, Ksrc, Kp);
    k_cvt_pad<<<1024, 256, 0, stream>>>(Wr, W16r, HH, Ksrc, Kp);

    launch_gemm(X2, W16z, bz, XZ, MROWS, HH, Kp, stream);
    launch_gemm(X2, W16h, bh, XH, MROWS, HH, Kp, stream);
    launch_gemm(X2, W16r, br, XR, MROWS, HH, Kp, stream);

    k_cvt_pad<<<512, 256, 0, stream>>>(Uz, U16z, HH, HH, HH);
    k_cvt_pad<<<512, 256, 0, stream>>>(Uh, U16h, HH, HH, HH);
    k_cvt_pad<<<512, 256, 0, stream>>>(Ur, U16r, HH, HH, HH);

    k_init_bar<<<1, 32, 0, stream>>>(BAR);
    k_gru_scan<<<SCAN_WGS, 64, 0, stream>>>(XZ, XH, XR, U16z, U16h, U16r, HS, H16, BAR);
  }

  // final classifier
  k_build_fcin<<<2048, 256, 0, stream>>>(HS, FCIN);
  k_cvt_pad<<<2048, 256, 0, stream>>>((const float*)d_in[21], FCW, NCLS, 1024, 1024);
  launch_gemm(FCIN, FCW, (const float*)d_in[22], OUT + 2, FCM, NCLS, 1024, stream);

  k_logsoftmax<<<FCM, 256, 0, stream>>>(OUT + 2, lab, NLLb, ERRb);
  k_reduce<<<1, 256, 0, stream>>>(NLLb, ERRb, OUT);
}
